// GINNet_2783138808359
// MI455X (gfx1250) — compile-verified, matrix operand order checked by reading, not run on hardware
//
#include <hip/hip_runtime.h>

#define N_NODES 50000
#define N_EDGES 800000
#define N_GRAPHS 512
#define IN_C 128
#define HID 256
#define OUT_C 16

typedef __attribute__((ext_vector_type(16))) __bf16 v16bf;
typedef __attribute__((ext_vector_type(8)))  float  v8f;

__device__ __forceinline__ unsigned short f2bf(float f) {
  union { float f; unsigned int u; } c; c.f = f;
  unsigned int u = c.u;
  u += 0x7fffu + ((u >> 16) & 1u);   // round-to-nearest-even
  return (unsigned short)(u >> 16);
}

// fragment K-offset for (lane, p) per documented 16-bit 16x32 WMMA layout:
// lanes 0-15 hold K {0..7, 16..23}; lanes 16-31 hold K {8..15, 24..31}; pairs packed per dword
__device__ __forceinline__ int frag_koff(int lane, int p) {
  int kb = (lane < 16) ? 0 : 8;
  return (p < 4) ? (kb + 2 * p) : (16 + kb + 2 * (p - 4));
}

// ---------------- agg init: dst <- src (float4) ----------------
__global__ void copy_f4(const float* __restrict__ src, float* __restrict__ dst, int n4) {
  int i = blockIdx.x * blockDim.x + threadIdx.x;
  if (i < n4) ((float4*)dst)[i] = ((const float4*)src)[i];
}

// ---------------- edge scatter-add: agg[dst] += h[src] ----------------
__global__ void edge_scatter(const float* __restrict__ h, const int* __restrict__ esrc,
                             const int* __restrict__ edst, float* __restrict__ agg,
                             int cg /* C/4, power of two */, long total) {
  long i = (long)blockIdx.x * blockDim.x + threadIdx.x;
  if (i >= total) return;
  int e = (int)(i / cg);
  int c = (int)(i - (long)e * cg);
  int s = esrc[e], d = edst[e];
  const float4 v = ((const float4*)(h + (long)s * (cg * 4)))[c];
  float* o = agg + (long)d * (cg * 4) + c * 4;
  atomicAdd(o + 0, v.x); atomicAdd(o + 1, v.y);
  atomicAdd(o + 2, v.z); atomicAdd(o + 3, v.w);
}

// ---------------- pack f32 activations [N x K] -> A-fragment order (bf16 pairs) ----------
// out uint index i = (rt*KT + kt)*256 + lane*8 + p ; value = {bf16(src[row][k]), bf16(src[row][k+1])}
__global__ void apack_f32(const float* __restrict__ src, unsigned int* __restrict__ out,
                          int K, int KT, long total) {
  long i = (long)blockIdx.x * blockDim.x + threadIdx.x;
  if (i >= total) return;
  int p = (int)(i & 7);
  int lane = (int)((i >> 3) & 31);
  long tt = i >> 8;
  int kt = (int)(tt % KT);
  int rt = (int)(tt / KT);
  int row = rt * 16 + (lane & 15);
  int k = kt * 32 + frag_koff(lane, p);
  const float* s = src + (long)row * K + k;
  out[i] = (unsigned int)f2bf(s[0]) | ((unsigned int)f2bf(s[1]) << 16);
}

// ---------------- pack bf16 activations [N x K] -> A-fragment order ----------------
__global__ void apack_bf16(const unsigned short* __restrict__ src, unsigned int* __restrict__ out,
                           int K, int KT, long total) {
  long i = (long)blockIdx.x * blockDim.x + threadIdx.x;
  if (i >= total) return;
  int p = (int)(i & 7);
  int lane = (int)((i >> 3) & 31);
  long tt = i >> 8;
  int kt = (int)(tt % KT);
  int rt = (int)(tt / KT);
  int row = rt * 16 + (lane & 15);
  int k = kt * 32 + frag_koff(lane, p);
  out[i] = *(const unsigned int*)(src + (long)row * K + k);   // (k, k+1) pair, 4B aligned
}

// ---------------- pack f32 weights [K x M] -> B-fragment order (cvt fused) ----------------
// out uint index i = (ct*KT + kt)*256 + lane*8 + p ; value = {bf16(W[k][col]), bf16(W[k+1][col])}
__global__ void wpack_f32(const float* __restrict__ W, unsigned int* __restrict__ out,
                          int K, int M, int KT, long total) {
  long i = (long)blockIdx.x * blockDim.x + threadIdx.x;
  if (i >= total) return;
  int p = (int)(i & 7);
  int lane = (int)((i >> 3) & 31);
  long tt = i >> 8;
  int kt = (int)(tt % KT);
  int ct = (int)(tt / KT);
  int col = ct * 16 + (lane & 15);
  int k = kt * 32 + frag_koff(lane, p);
  unsigned int lo = f2bf(W[(long)k * M + col]);
  unsigned int hi = f2bf(W[(long)(k + 1) * M + col]);
  out[i] = lo | (hi << 16);
}

// ---------------- WMMA GEMM on pre-packed fragments: no LDS, no barriers --------------
// 256 threads = 8 waves (4 row-tile-pairs x 2 col-tile-pairs); wave tile 32x32;
// block tile 128 rows x 64 cols. N, M multiples of 16; K multiple of 32.
template <int RELU, int OUTBF>
__global__ void gemm_pk(const uint4* __restrict__ Apk, const uint4* __restrict__ Wpk,
                        const float* __restrict__ bias, void* __restrict__ Cout,
                        int N, int K, int M) {
  const int tid  = threadIdx.x;
  const int lane = tid & 31;
  const int wave = tid >> 5;
  const int wr   = wave & 3;
  const int wc   = wave >> 2;
  const int KT   = K >> 5;
  const int nrt  = N >> 4;
  const int nct  = M >> 4;
  const int rt0  = blockIdx.y * 8 + wr * 2;
  const int ct0  = blockIdx.x * 4 + wc * 2;
  const bool r0 = rt0 < nrt,     r1 = (rt0 + 1) < nrt;
  const bool c0 = ct0 < nct,     c1 = (ct0 + 1) < nct;
  const int rA0 = r0 ? rt0 : 0,  rA1 = r1 ? (rt0 + 1) : 0;   // clamp to keep loads in-bounds
  const int cB0 = c0 ? ct0 : 0,  cB1 = c1 ? (ct0 + 1) : 0;

  // uint4 index for (tile, kt=0, lane): ((tile*KT)*32 + lane)*2 ; per-kt stride = 64 uint4
  long ia0 = ((long)rA0 * KT * 32 + lane) * 2;
  long ia1 = ((long)rA1 * KT * 32 + lane) * 2;
  long ib0 = ((long)cB0 * KT * 32 + lane) * 2;
  long ib1 = ((long)cB1 * KT * 32 + lane) * 2;
  const long step = 64;

  v8f acc00 = {}, acc01 = {}, acc10 = {}, acc11 = {};
  union F { uint4 q[2]; v16bf v; };

  for (int kt = 0; kt < KT; ++kt) {
    F a0, a1, b0, b1;
    a0.q[0] = Apk[ia0]; a0.q[1] = Apk[ia0 + 1];
    a1.q[0] = Apk[ia1]; a1.q[1] = Apk[ia1 + 1];
    b0.q[0] = Wpk[ib0]; b0.q[1] = Wpk[ib0 + 1];
    b1.q[0] = Wpk[ib1]; b1.q[1] = Wpk[ib1 + 1];
    if (kt + 1 < KT) {
      __builtin_prefetch((const void*)(Apk + ia0 + step), 0, 3);
      __builtin_prefetch((const void*)(Apk + ia1 + step), 0, 3);
    }
    ia0 += step; ia1 += step; ib0 += step; ib1 += step;
    acc00 = __builtin_amdgcn_wmma_f32_16x16x32_bf16(false, a0.v, false, b0.v, (short)0, acc00, false, false);
    acc01 = __builtin_amdgcn_wmma_f32_16x16x32_bf16(false, a0.v, false, b1.v, (short)0, acc01, false, false);
    acc10 = __builtin_amdgcn_wmma_f32_16x16x32_bf16(false, a1.v, false, b0.v, (short)0, acc10, false, false);
    acc11 = __builtin_amdgcn_wmma_f32_16x16x32_bf16(false, a1.v, false, b1.v, (short)0, acc11, false, false);
  }

  // epilogue: C layout VGPR r -> M=r (lanes 0-15) / M=r+8 (lanes 16-31), N = lane&15
  const int rowL = lane & 15;
  auto st = [&](const v8f& acc, int rt, bool rok, int ct, bool cok) {
    if (!rok || !cok) return;
    int col = ct * 16 + rowL;
    #pragma unroll
    for (int r = 0; r < 8; ++r) {
      int row = rt * 16 + ((lane < 16) ? r : r + 8);
      float o = acc[r] + bias[col];
      if (RELU) o = fmaxf(o, 0.0f);
      if (OUTBF) ((unsigned short*)Cout)[(long)row * M + col] = f2bf(o);
      else       ((float*)Cout)[(long)row * M + col] = o;
    }
  };
  st(acc00, rt0, r0, ct0, c0);         st(acc01, rt0, r0, ct0 + 1, c1);
  st(acc10, rt0 + 1, r1, ct0, c0);     st(acc11, rt0 + 1, r1, ct0 + 1, c1);
}

// ---------------- pooling ----------------
__global__ void count_nodes(const int* __restrict__ batch, float* __restrict__ counts) {
  int n = blockIdx.x * blockDim.x + threadIdx.x;
  if (n < N_NODES) atomicAdd(&counts[batch[n]], 1.0f);
}

__global__ void pool_sum(const float* __restrict__ h, const int* __restrict__ batch,
                         float* __restrict__ pooled) {
  const int cg = HID / 4;  // 64
  long i = (long)blockIdx.x * blockDim.x + threadIdx.x;
  if (i >= (long)N_NODES * cg) return;
  int n = (int)(i / cg);
  int c = (int)(i - (long)n * cg);
  int g = batch[n];
  float4 v = ((const float4*)(h + (long)n * HID))[c];
  float* o = pooled + (long)g * HID + c * 4;
  atomicAdd(o + 0, v.x); atomicAdd(o + 1, v.y);
  atomicAdd(o + 2, v.z); atomicAdd(o + 3, v.w);
}

__global__ void pool_div(float* pooled, const float* __restrict__ counts) {
  int i = blockIdx.x * blockDim.x + threadIdx.x;
  if (i < N_GRAPHS * HID) {
    float c = counts[i / HID];
    pooled[i] /= fmaxf(c, 1.0f);
  }
}

extern "C" void kernel_launch(void* const* d_in, const int* in_sizes, int n_in,
                              void* d_out, int out_size, void* d_ws, size_t ws_size,
                              hipStream_t stream) {
  const float* x     = (const float*)d_in[0];
  const int*   ei    = (const int*)d_in[1];
  const int*   esrc  = ei;
  const int*   edst  = ei + N_EDGES;
  const int*   batch = (const int*)d_in[2];
  const float* cw1[3] = {(const float*)d_in[3],  (const float*)d_in[7],  (const float*)d_in[11]};
  const float* cb1[3] = {(const float*)d_in[4],  (const float*)d_in[8],  (const float*)d_in[12]};
  const float* cw2[3] = {(const float*)d_in[5],  (const float*)d_in[9],  (const float*)d_in[13]};
  const float* cb2[3] = {(const float*)d_in[6],  (const float*)d_in[10], (const float*)d_in[14]};
  const float* ow1 = (const float*)d_in[15];
  const float* ob1 = (const float*)d_in[16];
  const float* ow2 = (const float*)d_in[17];
  const float* ob2 = (const float*)d_in[18];

  // ---- workspace carve-up (256B aligned bumps) ----
  char* ws = (char*)d_ws;
  size_t off = 0;
  auto bump = [&](size_t bytes) {
    void* p = ws + off;
    off += (bytes + 255) & ~(size_t)255;
    return p;
  };
  const int NRT = N_NODES / 16;   // 3125 (exact)
  float*          B0     = (float*)bump((size_t)N_NODES * HID * 4);           // agg (f32)
  float*          B2     = (float*)bump((size_t)N_NODES * HID * 4);           // h   (f32)
  unsigned int*   Apk    = (unsigned int*)bump((size_t)NRT * 8 * 256 * 4);    // packed A
  unsigned short* Mbf    = (unsigned short*)bump((size_t)N_NODES * HID * 2);  // mid row-major bf16
  unsigned int*   Mpk    = (unsigned int*)bump((size_t)NRT * 8 * 256 * 4);    // packed mid
  float*          pooled = (float*)bump((size_t)N_GRAPHS * HID * 4);
  float*          counts = (float*)bump((size_t)N_GRAPHS * 4);
  unsigned int*   Ppk    = (unsigned int*)bump((size_t)(N_GRAPHS / 16) * 8 * 256 * 4);
  unsigned short* MPbf   = (unsigned short*)bump((size_t)N_GRAPHS * HID * 2);
  unsigned int*   MPpk   = (unsigned int*)bump((size_t)(N_GRAPHS / 16) * 8 * 256 * 4);
  unsigned int*   wpk1[3], *wpk2[3];
  for (int l = 0; l < 3; ++l) {
    wpk1[l] = (unsigned int*)bump((size_t)(HID / 16) * 8 * 256 * 4);
    wpk2[l] = (unsigned int*)bump((size_t)(HID / 16) * 8 * 256 * 4);
  }
  unsigned int* owpk1 = (unsigned int*)bump((size_t)(HID / 16) * 8 * 256 * 4);
  unsigned int* owpk2 = (unsigned int*)bump((size_t)(OUT_C / 16) * 8 * 256 * 4);

  dim3 blk(256);
  auto wpack = [&](const float* W, unsigned int* dst, int K, int M) {
    int KT = K / 32;
    long total = (long)(M / 16) * KT * 256;
    wpack_f32<<<(unsigned)((total + 255) / 256), blk, 0, stream>>>(W, dst, K, M, KT, total);
  };

  // pack all weights once (cvt fused)
  wpack(cw1[0], wpk1[0], IN_C, HID);
  wpack(cw2[0], wpk2[0], HID, HID);
  for (int l = 1; l < 3; ++l) {
    wpack(cw1[l], wpk1[l], HID, HID);
    wpack(cw2[l], wpk2[l], HID, HID);
  }
  wpack(ow1, owpk1, HID, HID);
  wpack(ow2, owpk2, HID, OUT_C);

  const float* h = x;
  int C = IN_C;
  for (int l = 0; l < 3; ++l) {
    int n4 = N_NODES * C / 4;
    copy_f4<<<(n4 + 255) / 256, blk, 0, stream>>>(h, B0, n4);
    long tot = (long)N_EDGES * (C / 4);
    edge_scatter<<<(unsigned)((tot + 255) / 256), blk, 0, stream>>>(h, esrc, edst, B0, C / 4, tot);

    int KT = C / 32;
    long atot = (long)NRT * KT * 256;
    apack_f32<<<(unsigned)((atot + 255) / 256), blk, 0, stream>>>(B0, Apk, C, KT, atot);

    dim3 g1((HID / 16 + 3) / 4, (NRT + 7) / 8);
    gemm_pk<1, 1><<<g1, blk, 0, stream>>>((const uint4*)Apk, (const uint4*)wpk1[l], cb1[l],
                                          Mbf, N_NODES, C, HID);
    long mtot = (long)NRT * (HID / 32) * 256;
    apack_bf16<<<(unsigned)((mtot + 255) / 256), blk, 0, stream>>>(Mbf, Mpk, HID, HID / 32, mtot);
    dim3 g2((HID / 16 + 3) / 4, (NRT + 7) / 8);
    gemm_pk<1, 0><<<g2, blk, 0, stream>>>((const uint4*)Mpk, (const uint4*)wpk2[l], cb2[l],
                                          B2, N_NODES, HID, HID);
    h = B2;
    C = HID;
  }

  // pooling
  hipMemsetAsync(pooled, 0, (size_t)N_GRAPHS * HID * 4, stream);
  hipMemsetAsync(counts, 0, (size_t)N_GRAPHS * 4, stream);
  count_nodes<<<(N_NODES + 255) / 256, blk, 0, stream>>>(batch, counts);
  long ptot = (long)N_NODES * (HID / 4);
  pool_sum<<<(unsigned)((ptot + 255) / 256), blk, 0, stream>>>(B2, batch, pooled);
  pool_div<<<(N_GRAPHS * HID + 255) / 256, blk, 0, stream>>>(pooled, counts);

  // output MLP (N = 512, multiple of 16)
  const int GRT = N_GRAPHS / 16;  // 32
  long gtot = (long)GRT * (HID / 32) * 256;
  apack_f32<<<(unsigned)((gtot + 255) / 256), blk, 0, stream>>>(pooled, Ppk, HID, HID / 32, gtot);
  dim3 g3((HID / 16 + 3) / 4, (GRT + 7) / 8);
  gemm_pk<1, 1><<<g3, blk, 0, stream>>>((const uint4*)Ppk, (const uint4*)owpk1, ob1,
                                        MPbf, N_GRAPHS, HID, HID);
  apack_bf16<<<(unsigned)((gtot + 255) / 256), blk, 0, stream>>>(MPbf, MPpk, HID, HID / 32, gtot);
  dim3 g4((OUT_C / 16 + 3) / 4, (GRT + 7) / 8);
  gemm_pk<0, 0><<<g4, blk, 0, stream>>>((const uint4*)MPpk, (const uint4*)owpk2, ob2,
                                        (float*)d_out, N_GRAPHS, HID, OUT_C);
}
